// SyntaxErrorGNN_68307159876136
// MI455X (gfx1250) — matrix operand order repaired and checked by reading, not compiled, not run on hardware
//
#include <hip/hip_runtime.h>

// ---------------- problem constants (match reference) ----------------
#define NN   100000     // nodes
#define EE   1600000    // edges
#define NG   256        // graphs
#define HH   64         // hidden / input dim
#define CC   6          // classes
#define BN_EPS 1e-5f

typedef _Float16 v16h __attribute__((ext_vector_type(16)));
typedef _Float16 v8h  __attribute__((ext_vector_type(8)));
typedef float    v8f  __attribute__((ext_vector_type(8)));

// ---------------- degree / edge norm ----------------
__global__ void deg_init_kernel(float* deg) {
  int i = blockIdx.x * 256 + threadIdx.x;
  if (i < NN) deg[i] = 1.0f;                       // self-loop included
}

__global__ void deg_accum_kernel(const int* __restrict__ dst, float* deg) {
  int e = blockIdx.x * 256 + threadIdx.x;
  if (e < EE) atomicAdd(&deg[dst[e]], 1.0f);
}

__global__ void edge_norm_kernel(const int* __restrict__ src, const int* __restrict__ dst,
                                 const float* __restrict__ deg, float* __restrict__ norm_e) {
  int e = blockIdx.x * 256 + threadIdx.x;
  if (e < EE) norm_e[e] = rsqrtf(deg[src[e]]) * rsqrtf(deg[dst[e]]);
}

// ---------------- f16 conversions / weight pre-pack ----------------
__global__ void xconv_kernel(const float* __restrict__ x, _Float16* __restrict__ xh) {
  int i = blockIdx.x * 256 + threadIdx.x;
  if (i < NN * HH) xh[i] = (_Float16)x[i];
}

// Pack W[64x64] (row-major, k-major) into WMMA B-operand lane order:
// index = ((colTile*2 + ksIdx)*32 + lane)*16 + t ; element = W[k][n]
//   n = colTile*16 + (lane&15), k = ksIdx*32 + (lane>=16 ? 16 : 0) + t
__global__ void wconv_kernel(const float* __restrict__ W1, const float* __restrict__ W2,
                             const float* __restrict__ W3, _Float16* __restrict__ WB) {
  int i = blockIdx.x * 256 + threadIdx.x;
  if (i >= 3 * 4096) return;
  int w = i >> 12;
  int r = i & 4095;
  int t       = r & 15;
  int lane    = (r >> 4) & 31;
  int ksIdx   = (r >> 9) & 1;
  int colTile = (r >> 10) & 3;
  int n = colTile * 16 + (lane & 15);
  int k = ksIdx * 32 + ((lane >= 16) ? 16 : 0) + t;
  const float* W = (w == 0) ? W1 : (w == 1) ? W2 : W3;
  WB[i] = (_Float16)W[k * HH + n];
}

// ---------------- dense matmul: Y[N,64] = A[N,64] (f16) @ W (pre-packed f16), f32 accum ----------------
// 256 threads = 8 waves; wave w -> rowTile = blockIdx.x*2 + (w>>2), colTile = w&3
__global__ void gemm64_wmma_kernel(const _Float16* __restrict__ A,
                                   const _Float16* __restrict__ WB,
                                   float* __restrict__ Y) {
  const int lane = threadIdx.x & 31;
  const int wave = threadIdx.x >> 5;
  const int rowTile = blockIdx.x * 2 + (wave >> 2);
  const int colTile = wave & 3;
  const int m  = lane & 15;
  const int hi = (lane >= 16) ? 1 : 0;

  v8f acc = {};
#pragma unroll
  for (int ksIdx = 0; ksIdx < 2; ++ksIdx) {
    // A lane data: halves 0..15 = K[base..base+15], base = ksIdx*32 + hi*16
    const _Float16* arow = A + (size_t)(rowTile * 16 + m) * HH + ksIdx * 32 + hi * 16;
    v8h a0 = *(const v8h*)(arow);
    v8h a1 = *(const v8h*)(arow + 8);
    v16h av = __builtin_shufflevector(a0, a1, 0,1,2,3,4,5,6,7,8,9,10,11,12,13,14,15);
    // B lane data: one contiguous 32B chunk from the pre-packed weight
    v16h bv = *(const v16h*)(WB + (((size_t)colTile * 2 + ksIdx) * 32 + lane) * 16);
    acc = __builtin_amdgcn_wmma_f32_16x16x32_f16(false, av, false, bv,
                                                 (short)0, acc, false, false);
  }
  // D layout: VGPR v -> row (v + 8*hi), col = m
  float* yrow = Y + ((size_t)rowTile * 16 + hi * 8) * HH + colTile * 16 + m;
#pragma unroll
  for (int v = 0; v < 8; ++v) yrow[(size_t)v * HH] = acc[v];
}

// ---------------- GCN aggregation ----------------
__global__ void self_term_kernel(const float* __restrict__ hlin, const float* __restrict__ deg,
                                 const float* __restrict__ bias, float* __restrict__ agg) {
  int i = blockIdx.x * 256 + threadIdx.x;
  if (i >= NN * HH) return;
  int n = i >> 6, c = i & 63;
  agg[i] = hlin[i] * (1.0f / deg[n]) + bias[c];
}

// one thread per (edge, 4 channels): float4 gather from L2-resident hlin, 4 f32 atomics
__global__ void edge_scatter_kernel(const int* __restrict__ src, const int* __restrict__ dst,
                                    const float* __restrict__ norm_e,
                                    const float* __restrict__ hlin, float* __restrict__ agg) {
  int i = blockIdx.x * 256 + threadIdx.x;
  if (i >= EE * 16) return;
  int e = i >> 4, q = i & 15;
  int s = src[e], d = dst[e];
  float w = norm_e[e];
  const float4 v = ((const float4*)hlin)[(size_t)s * 16 + q];
  float* ap = agg + (size_t)d * HH + q * 4;
  atomicAdd(ap + 0, v.x * w);
  atomicAdd(ap + 1, v.y * w);
  atomicAdd(ap + 2, v.z * w);
  atomicAdd(ap + 3, v.w * w);
}

// ---------------- batch norm ----------------
__global__ void zero_f_kernel(float* p, int n) {
  int i = blockIdx.x * 256 + threadIdx.x;
  if (i < n) p[i] = 0.0f;
}

__global__ void bn_stats_kernel(const float* __restrict__ agg, float* __restrict__ stats) {
  __shared__ float sS[256], sQ[256];
  int c = threadIdx.x & 63;
  int g = threadIdx.x >> 6;                      // 0..3
  float s = 0.f, q = 0.f;
  for (int n = blockIdx.x * 4 + g; n < NN; n += gridDim.x * 4) {
    float v = agg[(size_t)n * HH + c];
    s += v; q += v * v;
  }
  sS[threadIdx.x] = s; sQ[threadIdx.x] = q;
  __syncthreads();
  if (threadIdx.x < 64) {
    s = sS[c] + sS[c + 64] + sS[c + 128] + sS[c + 192];
    q = sQ[c] + sQ[c + 64] + sQ[c + 128] + sQ[c + 192];
    atomicAdd(&stats[c], s);
    atomicAdd(&stats[64 + c], q);
  }
}

__global__ void bn_finalize_kernel(const float* __restrict__ stats, const float* __restrict__ g,
                                   const float* __restrict__ be, float* __restrict__ param) {
  int c = threadIdx.x;
  if (c >= 64) return;
  float mean = stats[c] * (1.0f / NN);
  float var  = stats[64 + c] * (1.0f / NN) - mean * mean;
  float sc   = g[c] * rsqrtf(var + BN_EPS);
  param[c]      = sc;
  param[64 + c] = be[c] - mean * sc;
}

__global__ void bn_apply_kernel(const float* __restrict__ agg, const float* __restrict__ param,
                                _Float16* __restrict__ actH, float* __restrict__ actF) {
  int i = blockIdx.x * 256 + threadIdx.x;
  if (i >= NN * HH) return;
  int c = i & 63;
  float v = fmaxf(agg[i] * param[c] + param[64 + c], 0.0f);
  actH[i] = (_Float16)v;
  actF[i] = v;
}

// ---------------- pooling ----------------
__global__ void pool_init_kernel(float* gsum, unsigned* gmaxb, float* gcnt) {
  int i = blockIdx.x * 256 + threadIdx.x;
  if (i < NG * HH) { gsum[i] = 0.0f; gmaxb[i] = 0u; }   // post-ReLU values >= 0 -> max init 0 == ref
  if (i < NG) gcnt[i] = 0.0f;
}

__global__ void pool_cnt_kernel(const int* __restrict__ batch, float* gcnt) {
  int n = blockIdx.x * 256 + threadIdx.x;
  if (n < NN) atomicAdd(&gcnt[batch[n]], 1.0f);
}

__global__ void pool_accum_kernel(const float* __restrict__ act, const int* __restrict__ batch,
                                  float* __restrict__ gsum, unsigned* __restrict__ gmaxb) {
  int i = blockIdx.x * 256 + threadIdx.x;
  if (i >= NN * HH) return;
  int n = i >> 6, c = i & 63;
  int b = batch[n];
  float v = act[i];
  atomicAdd(&gsum[b * HH + c], v);
  atomicMax(&gmaxb[b * HH + c], __float_as_uint(v));    // valid ordering: v >= 0
}

__global__ void pool_final_kernel(const float* __restrict__ gsum, const unsigned* __restrict__ gmaxb,
                                  const float* __restrict__ gcnt, float* __restrict__ z) {
  int i = blockIdx.x * 256 + threadIdx.x;
  if (i >= NG * HH) return;
  int b = i >> 6, c = i & 63;
  float cnt = fmaxf(gcnt[b], 1.0f);
  z[b * 128 + c]      = gsum[i] / cnt;
  z[b * 128 + 64 + c] = __uint_as_float(gmaxb[i]);
}

// ---------------- MLP head: one block (128 threads) per graph ----------------
__global__ void mlp_head_kernel(const float* __restrict__ z,
                                const float* __restrict__ LW1, const float* __restrict__ Lb1,
                                const float* __restrict__ LW2, const float* __restrict__ Lb2,
                                const float* __restrict__ LW3, const float* __restrict__ Lb3,
                                float* __restrict__ out) {
  __shared__ float zz[128], t1[64], t2[32];
  int b = blockIdx.x, t = threadIdx.x;
  zz[t] = z[b * 128 + t];
  __syncthreads();
  if (t < 64) {
    float s = Lb1[t];
    for (int k = 0; k < 128; ++k) s += zz[k] * LW1[k * 64 + t];
    t1[t] = fmaxf(s, 0.0f);
  }
  __syncthreads();
  if (t < 32) {
    float s = Lb2[t];
    for (int k = 0; k < 64; ++k) s += t1[k] * LW2[k * 32 + t];
    t2[t] = fmaxf(s, 0.0f);
  }
  __syncthreads();
  if (t < CC) {
    float s = Lb3[t];
    for (int k = 0; k < 32; ++k) s += t2[k] * LW3[k * CC + t];
    out[b * CC + t] = s;
  }
}

// ---------------- host launch ----------------
#define BLK(n) dim3(((n) + 255) / 256), dim3(256)

extern "C" void kernel_launch(void* const* d_in, const int* in_sizes, int n_in,
                              void* d_out, int out_size, void* d_ws, size_t ws_size,
                              hipStream_t stream) {
  const float* x     = (const float*)d_in[0];
  const int*   src   = (const int*)d_in[1];
  const int*   dst   = (const int*)d_in[2];
  const int*   batch = (const int*)d_in[3];
  const float* Wl[3] = {(const float*)d_in[4], (const float*)d_in[6], (const float*)d_in[8]};
  const float* bl[3] = {(const float*)d_in[5], (const float*)d_in[7], (const float*)d_in[9]};
  const float* gl[3] = {(const float*)d_in[10], (const float*)d_in[12], (const float*)d_in[14]};
  const float* bel[3]= {(const float*)d_in[11], (const float*)d_in[13], (const float*)d_in[15]};
  const float* LW1 = (const float*)d_in[16]; const float* Lb1 = (const float*)d_in[17];
  const float* LW2 = (const float*)d_in[18]; const float* Lb2 = (const float*)d_in[19];
  const float* LW3 = (const float*)d_in[20]; const float* Lb3 = (const float*)d_in[21];
  float* out = (float*)d_out;

  // ---- carve workspace ----
  char* base = (char*)d_ws;
  size_t off = 0;
  auto carve = [&](size_t bytes) -> char* {
    char* p = base + off;
    off += (bytes + 255) & ~(size_t)255;
    return p;
  };
  float*     deg    = (float*)carve((size_t)NN * 4);
  float*     norm_e = (float*)carve((size_t)EE * 4);
  _Float16*  WB     = (_Float16*)carve((size_t)3 * 4096 * 2);
  _Float16*  actH   = (_Float16*)carve((size_t)NN * HH * 2);   // f16 activations (matmul A)
  float*     hlin   = (float*)carve((size_t)NN * HH * 4);      // matmul out; reused as f32 act
  float*     agg    = (float*)carve((size_t)NN * HH * 4);
  float*     stats  = (float*)carve(128 * 4);                  // sums | sumsq
  float*     param  = (float*)carve(128 * 4);                  // scale | shift
  float*     gsum   = (float*)carve((size_t)NG * HH * 4);
  unsigned*  gmaxb  = (unsigned*)carve((size_t)NG * HH * 4);
  float*     gcnt   = (float*)carve((size_t)NG * 4);
  float*     zbuf   = (float*)carve((size_t)NG * 128 * 4);
  (void)ws_size; (void)n_in; (void)in_sizes; (void)out_size;

  // ---- degree + edge norms ----
  deg_init_kernel<<<BLK(NN), 0, stream>>>(deg);
  deg_accum_kernel<<<BLK(EE), 0, stream>>>(dst, deg);
  edge_norm_kernel<<<BLK(EE), 0, stream>>>(src, dst, deg, norm_e);

  // ---- one-time conversions ----
  wconv_kernel<<<BLK(3 * 4096), 0, stream>>>(Wl[0], Wl[1], Wl[2], WB);
  xconv_kernel<<<BLK(NN * HH), 0, stream>>>(x, actH);

  // ---- three GCN layers ----
  for (int l = 0; l < 3; ++l) {
    gemm64_wmma_kernel<<<dim3(NN / 32), dim3(256), 0, stream>>>(actH, WB + (size_t)l * 4096, hlin);
    self_term_kernel<<<BLK(NN * HH), 0, stream>>>(hlin, deg, bl[l], agg);
    edge_scatter_kernel<<<BLK(EE * 16), 0, stream>>>(src, dst, norm_e, hlin, agg);
    zero_f_kernel<<<dim3(1), dim3(256), 0, stream>>>(stats, 128);
    bn_stats_kernel<<<dim3(256), dim3(256), 0, stream>>>(agg, stats);
    bn_finalize_kernel<<<dim3(1), dim3(64), 0, stream>>>(stats, gl[l], bel[l], param);
    bn_apply_kernel<<<BLK(NN * HH), 0, stream>>>(agg, param, actH, hlin);  // hlin := f32 act
  }

  // ---- pooling ----
  pool_init_kernel<<<BLK(NG * HH), 0, stream>>>(gsum, gmaxb, gcnt);
  pool_cnt_kernel<<<BLK(NN), 0, stream>>>(batch, gcnt);
  pool_accum_kernel<<<BLK(NN * HH), 0, stream>>>(hlin, batch, gsum, gmaxb);
  pool_final_kernel<<<BLK(NG * HH), 0, stream>>>(gsum, gmaxb, gcnt, zbuf);

  // ---- MLP head ----
  mlp_head_kernel<<<dim3(NG), dim3(128), 0, stream>>>(zbuf, LW1, Lb1, LW2, Lb2, LW3, Lb3, out);
}
